// SuperpositionTTLayer_67216238182705
// MI455X (gfx1250) — matrix operand order repaired and checked by reading, not compiled, not run on hardware
//
#include <hip/hip_runtime.h>

typedef __attribute__((ext_vector_type(16))) __bf16 v16bf;
typedef __attribute__((ext_vector_type(8)))  float  v8f;

#define N_TOK       4096
#define TOK_PER_BLK 8
#define GRID_BLKS   (N_TOK / TOK_PER_BLK)   // 512
#define THREADS     256                      // 8 wave32 waves
#define NW          (THREADS / 32)

// ---- LDS layout (bf16 element offsets) ----
#define OFF_C0 0                         // [s=4][j=64][k=32]  col-major (j,k); k=i0 zero-padded 8->32
#define SZ_C0  (4 * 64 * 32)
#define OFF_C1 (OFF_C0 + SZ_C0)          // [s=4][j=128][k=128] col-major; j=o1*8+r2, k=i1*8+r1
#define SZ_C1  (4 * 128 * 128)
#define OFF_C2 (OFF_C1 + SZ_C1)          // [s=4][j=16][k=64]  col-major; j=o2 zero-padded 8->16, k=i2*8+r2
#define SZ_C2  (4 * 16 * 64)
#define OFF_X  (OFF_C2 + SZ_C2)          // [m=128][k=32] row-major; m=i1*8+i2, k=i0 zero-padded
#define SZ_X   (128 * 32)
#define OFF_T1 (OFF_X + SZ_X)            // [m=128][j=64] row-major; m=(i1,i2), j=(o0,r1)
#define SZ_T1  (128 * 64)
#define OFF_T2 (OFF_T1 + SZ_T1)          // [m=64][j=128] row-major; m=(o0,i2), j=(o1,r2)
#define SZ_T2  (64 * 128)
#define OFF_OUT (OFF_T2 + SZ_T2)         // f32 [1024][4] staged output (8192 bf16-els = 16 KB)
#define SZ_OUT  (4096 * 2)               // in bf16 elements
#define SMEM_ELEMS (OFF_OUT + SZ_OUT)    // 106496 el = 208 KB

__device__ __forceinline__ v8f wmma_bf16(v16bf a, v16bf b, v8f c) {
  // 8 args: (neg_a, A, neg_b, B, c_mod, C, reuse_a, reuse_b)
  return __builtin_amdgcn_wmma_f32_16x16x32_bf16(false, a, false, b, (short)0, c,
                                                 false, false);
}

// A tile (16x32 bf16) from row-major LDS (row stride lda), origin = base.
// A layout: lanes 0-15 / 16-31 both hold rows M=0..15; VGPR v: K = 2*(v&3) + 8*g + 16*(v>>2).
__device__ __forceinline__ v16bf load_a_rm(const __bf16* base, int lda, int lane) {
  const int g = (lane >> 4) & 1;
  const int m = lane & 15;
  const __bf16* rp = base + m * lda;
  v16bf a;
#pragma unroll
  for (int v = 0; v < 8; ++v) {
    const int k = 2 * (v & 3) + 8 * g + 16 * (v >> 2);
    a[2 * v]     = rp[k];
    a[2 * v + 1] = rp[k + 1];
  }
  return a;
}

// B tile (32x16 bf16) from col-major LDS (col stride ldk), origin = base (k=0 of tile).
// B layout: lane N = lane&15; VGPR v: K = 2*(v&3) + 8*g + 16*(v>>2) (matches A).
__device__ __forceinline__ v16bf load_b_cm(const __bf16* base, int ldk, int lane) {
  const int g = (lane >> 4) & 1;
  const int n = lane & 15;
  const __bf16* cp = base + n * ldk;
  v16bf b;
#pragma unroll
  for (int v = 0; v < 8; ++v) {
    const int k = 2 * (v & 3) + 8 * g + 16 * (v >> 2);
    b[2 * v]     = cp[k];
    b[2 * v + 1] = cp[k + 1];
  }
  return b;
}

// C/D layout: VGPR r -> row r + 8*g, col = lane&15. Store as bf16, row-major stride ldt.
__device__ __forceinline__ void store_c_bf16(__bf16* base, int ldt, int lane, v8f c) {
  const int g = (lane >> 4) & 1;
  const int col = lane & 15;
#pragma unroll
  for (int r = 0; r < 8; ++r)
    base[(r + 8 * g) * ldt + col] = (__bf16)c[r];
}

__global__ void __launch_bounds__(THREADS)
tt_superposition_kernel(const float* __restrict__ x,
                        const float* __restrict__ core0,
                        const float* __restrict__ core1,
                        const float* __restrict__ core2,
                        float* __restrict__ out) {
  extern __shared__ __align__(16) __bf16 sm[];
  __bf16* c0L = sm + OFF_C0;
  __bf16* c1L = sm + OFF_C1;
  __bf16* c2L = sm + OFF_C2;
  __bf16* xL  = sm + OFF_X;
  __bf16* t1L = sm + OFF_T1;
  __bf16* t2L = sm + OFF_T2;
  float*  oL  = (float*)(sm + OFF_OUT);   // [1024][4] f32, 16B-aligned offset

  const int tid  = threadIdx.x;
  const int lane = tid & 31;
  const int wv   = tid >> 5;   // wave id 0..7 (uniform per wave)

  // ---------- stage cores once per block: fp32 -> bf16, rearranged ----------
  // core0 flat: [i0][o0][r1][s]  (leading rank-1 squeezed)
  for (int idx = tid; idx < SZ_C0; idx += THREADS) {
    const int s = idx >> 11;
    const int j = (idx >> 5) & 63;   // o0*8 + r1
    const int k = idx & 31;          // i0 (padded)
    float v = 0.f;
    if (k < 8) v = core0[((k * 8 + (j >> 3)) * 8 + (j & 7)) * 4 + s];
    c0L[idx] = (__bf16)v;
  }
  // core1 flat: [r1][i1][o1][r2][s]
  for (int idx = tid; idx < SZ_C1; idx += THREADS) {
    const int s = idx >> 14;
    const int j = (idx >> 7) & 127;  // o1*8 + r2
    const int k = idx & 127;         // i1*8 + r1
    const int o1 = j >> 3, r2 = j & 7;
    const int i1 = k >> 3, r1 = k & 7;
    c1L[idx] = (__bf16)core1[(((r1 * 16 + i1) * 16 + o1) * 8 + r2) * 4 + s];
  }
  // core2 flat: [r2][i2][o2][1][s]
  for (int idx = tid; idx < SZ_C2; idx += THREADS) {
    const int s = idx >> 10;
    const int j = (idx >> 6) & 15;   // o2 (padded to 16)
    const int k = idx & 63;          // i2*8 + r2
    float v = 0.f;
    if (j < 8) v = core2[(((k & 7) * 8 + (k >> 3)) * 8 + j) * 4 + s];
    c2L[idx] = (__bf16)v;
  }
  __syncthreads();

  const int n0 = blockIdx.x * TOK_PER_BLK;
  for (int t = 0; t < TOK_PER_BLK; ++t) {
    const int n = n0 + t;
    // stage token x: xL[m][k] = (k<8) ? x[n, i0=k, m=(i1*8+i2)] : 0
    const float* xp = x + (size_t)n * 1024;
    for (int idx = tid; idx < SZ_X; idx += THREADS) {
      const int m = idx >> 5, k = idx & 31;
      xL[idx] = (__bf16)((k < 8) ? xp[k * 128 + m] : 0.f);
    }
    if (t + 1 < TOK_PER_BLK)  // pull next token toward L2/WGP$ (global_prefetch_b8)
      __builtin_prefetch(x + (size_t)(n + 1) * 1024 + tid * 4, 0, 0);
    __syncthreads();

    for (int s = 0; s < 4; ++s) {
      const __bf16* c0s = c0L + s * (64 * 32);
      const __bf16* c1s = c1L + s * (128 * 128);
      const __bf16* c2s = c2L + s * (16 * 64);

      // ---- step 1: T1[128x64] = X[128xK32] * C0s[32x64]  (K padded 8->32) ----
#pragma unroll
      for (int it = 0; it < 32 / NW; ++it) {
        const int tile = it * NW + wv;       // uniform trip count -> scalar loop
        const int mt = tile >> 2, jt = tile & 3;
        v8f acc = {};
        v16bf a = load_a_rm(xL + mt * 16 * 32, 32, lane);
        v16bf b = load_b_cm(c0s + (jt * 16) * 32, 32, lane);
        acc = wmma_bf16(a, b, acc);
        store_c_bf16(t1L + (mt * 16) * 64 + jt * 16, 64, lane, acc);
      }
      __syncthreads();

      // ---- step 2: T2[64x128] = L2[64x128] * C1s[128x128], K = 4 chunks of 32 ----
      // L2[(o0,i2)][(i1,r1)] = T1[(i1*8+i2)][(o0*8+r1)]  (remap at A-load; pairs stay contiguous)
#pragma unroll
      for (int it = 0; it < 32 / NW; ++it) {
        const int tile = it * NW + wv;
        const int mt = tile >> 3, jt = tile & 7;
        const int g = (lane >> 4) & 1;
        const int m2 = mt * 16 + (lane & 15);
        const int o0 = m2 >> 3, i2 = m2 & 7;
        v8f acc = {};
#pragma unroll
        for (int kt = 0; kt < 4; ++kt) {
          v16bf a;
#pragma unroll
          for (int v = 0; v < 8; ++v) {
            const int k2 = kt * 32 + 2 * (v & 3) + 8 * g + 16 * (v >> 2); // even
            const int i1 = k2 >> 3, r1 = k2 & 7;
            const __bf16* p = t1L + (i1 * 8 + i2) * 64 + o0 * 8 + r1;
            a[2 * v]     = p[0];
            a[2 * v + 1] = p[1];
          }
          v16bf b = load_b_cm(c1s + (jt * 16) * 128 + kt * 32, 128, lane);
          acc = wmma_bf16(a, b, acc);
        }
        store_c_bf16(t2L + (mt * 16) * 128 + jt * 16, 128, lane, acc);
      }
      __syncthreads();

      // ---- step 3: OUT[128x8] = L3[128x64] * C2s[64x16pad], K = 2 chunks ----
      // L3[(o0,o1)][(i2,r2)] = T2[(o0*8+i2)][(o1*8+r2)] ; result staged to LDS f32 [1024][4]
      {
        const int mt = wv;                   // 8 tiles, 8 waves
        const int g = (lane >> 4) & 1;
        const int m3 = mt * 16 + (lane & 15);
        const int o0 = m3 >> 4, o1 = m3 & 15;
        v8f acc = {};
#pragma unroll
        for (int kt = 0; kt < 2; ++kt) {
          v16bf a;
#pragma unroll
          for (int v = 0; v < 8; ++v) {
            const int k3 = kt * 32 + 2 * (v & 3) + 8 * g + 16 * (v >> 2); // even
            const int i2 = k3 >> 3, r2 = k3 & 7;
            const __bf16* p = t2L + (o0 * 8 + i2) * 128 + o1 * 8 + r2;
            a[2 * v]     = p[0];
            a[2 * v + 1] = p[1];
          }
          v16bf b = load_b_cm(c2s + kt * 32, 64, lane);
          acc = wmma_bf16(a, b, acc);
        }
        const int col = lane & 15;           // o2; cols 8..15 are padding
        if (col < 8) {
#pragma unroll
          for (int r = 0; r < 8; ++r) {
            const int row = mt * 16 + r + 8 * g;      // (o0,o1) flat row
            oL[(row * 8 + col) * 4 + s] = acc[r];     // [d][s] scatter in LDS (cheap)
          }
        }
      }
      __syncthreads();
    }

    // ---- coalesced writeback: out[n, 0..1023, 0..3] as contiguous float4 ----
    {
      float4* dst = (float4*)(out + (size_t)n * 4096);
      const float4* src = (const float4*)oL;
#pragma unroll
      for (int i = 0; i < 1024 / THREADS; ++i)
        dst[i * THREADS + tid] = src[i * THREADS + tid];
    }
    __syncthreads();   // oL / xL reuse safety for next token
  }
}

extern "C" void kernel_launch(void* const* d_in, const int* in_sizes, int n_in,
                              void* d_out, int out_size, void* d_ws, size_t ws_size,
                              hipStream_t stream) {
  (void)in_sizes; (void)n_in; (void)out_size; (void)d_ws; (void)ws_size;
  const float* x  = (const float*)d_in[0];
  const float* c0 = (const float*)d_in[1];
  const float* c1 = (const float*)d_in[2];
  const float* c2 = (const float*)d_in[3];
  float* out = (float*)d_out;
  const size_t shmem = (size_t)SMEM_ELEMS * 2;  // 208 KB dynamic LDS
  tt_superposition_kernel<<<dim3(GRID_BLKS), dim3(THREADS), shmem, stream>>>(
      x, c0, c1, c2, out);
}
